// LinearAttention_24970939859449
// MI455X (gfx1250) — matrix operand order repaired and checked by reading, not compiled, not run on hardware
//
#include <hip/hip_runtime.h>
#include <hip/hip_bf16.h>
#include <cstdint>

// ---------------- CDNA5 WMMA types ----------------
typedef __attribute__((ext_vector_type(16))) __bf16 bf16x16;
typedef __attribute__((ext_vector_type(8)))  __bf16 bf16x8;
typedef __attribute__((ext_vector_type(8)))  float  f32x8;
typedef __attribute__((ext_vector_type(4)))  unsigned int u32x4;
typedef __attribute__((ext_vector_type(8)))  int i32x8;
typedef __attribute__((ext_vector_type(4)))  int i32x4;

#define WMMA_BF16(a, b, c) \
  __builtin_amdgcn_wmma_f32_16x16x32_bf16(false, (a), false, (b), (short)0, (c), false, false)

#ifndef __has_builtin
#define __has_builtin(x) 0
#endif
#if __has_builtin(__builtin_amdgcn_tensor_load_to_lds) && __has_builtin(__builtin_amdgcn_s_wait_tensorcnt)
#define HAVE_TDM 1
#else
#define HAVE_TDM 0
#endif

// ---------------- problem constants ----------------
static constexpr int B_   = 4;
static constexpr int L_   = 2048;
static constexpr int DM   = 1024;     // d_model
static constexpr int NH   = 16;       // heads
static constexpr int FD   = 16;       // feature dim per head (q/k)
static constexpr int HD   = 64;       // head dim (v)
static constexpr int FVAL = 273;      // 1 + 16 + 256 taylor features
static constexpr int FP_  = 288;      // padded to 18*16 for WMMA K
static constexpr int CHUNK = 64;      // causal chunk length
static constexpr int NCH  = L_ / CHUNK;

// ---------------- fragment loaders (wave32 layouts per ISA 7.12.2) ----------------
// A 16x32 bf16: lane row = M=lane%16; lanes0-15 K={0..7,16..23}, lanes16-31 K={8..15,24..31}
__device__ __forceinline__ bf16x16 lda_bf(const __bf16* p /* row base + k + ((lane>>4)<<3) */) {
  bf16x8 lo = *(const bf16x8*)(p);
  bf16x8 hi = *(const bf16x8*)(p + 16);
  return __builtin_shufflevector(lo, hi, 0,1,2,3,4,5,6,7,8,9,10,11,12,13,14,15);
}
// B 32x16 from row-major [N,K] bf16 source ("weight style"): N=lane%16; lanes0-15 K={0..7,16..23}, lanes16-31 K={8..15,24..31}
__device__ __forceinline__ bf16x16 ldb_w(const __bf16* base, int ld, int n, int k, int lane) {
  const __bf16* p = base + (size_t)(n + (lane & 15)) * ld + k + ((lane >> 4) << 3);
  return lda_bf(p);
}

// ---------------- TDM: 2D fp32 tile (rows x width, row stride) global -> LDS ----------------
#if HAVE_TDM
__device__ __forceinline__ void tdm_load_tile2d_f32(uint32_t lds_byte_off, uint64_t gaddr,
                                                    int width_elems, int rows, int stride_elems) {
  u32x4 g0;
  g0[0] = 1u;                                        // count=1, user descriptor
  g0[1] = lds_byte_off;                              // lds_addr
  g0[2] = (uint32_t)gaddr;                           // global_addr[31:0]
  g0[3] = (uint32_t)(gaddr >> 32) | (2u << 30);      // global_addr[56:32] | type=2
  i32x8 g1;
  g1[0] = (2 << 16);                                 // data_size = 4 bytes
  g1[1] = (int)((uint32_t)width_elems << 16);        // tensor_dim0[15:0]
  g1[2] = (int)((uint32_t)rows << 16);               // tensor_dim0[31:16]=0 | tensor_dim1[15:0]
  g1[3] = (int)((uint32_t)width_elems << 16);        // tensor_dim1[31:16]=0 | tile_dim0
  g1[4] = rows & 0xffff;                             // tile_dim1 | tile_dim2=0
  g1[5] = stride_elems;                              // tensor_dim0_stride[31:0]
  g1[6] = 0;                                         // stride hi | dim1_stride lo
  g1[7] = 0;
  i32x4 z4 = {0, 0, 0, 0};
#if __clang_major__ >= 23
  i32x8 z8 = {0, 0, 0, 0, 0, 0, 0, 0};
  __builtin_amdgcn_tensor_load_to_lds(g0, g1, z4, z4, z8, 0);
#else
  __builtin_amdgcn_tensor_load_to_lds(g0, g1, z4, z4, 0);
#endif
}
#endif

// ---------------- kernels ----------------
__global__ void cvt_f32_bf16(const float* __restrict__ src, __bf16* __restrict__ dst, int n) {
  int i = blockIdx.x * blockDim.x + threadIdx.x;
  if (i < n) dst[i] = (__bf16)src[i];
}

// C[m,n] = sum_k A[m,k] * W[n,k]  (x @ W.T), A/W bf16 row-major, out fp32 or bf16.
// One wave computes a 16(M) x 64(N) tile; 8 waves / block.
__global__ void __launch_bounds__(256)
gemm_wmma(const __bf16* __restrict__ A, const __bf16* __restrict__ W,
          float* __restrict__ Cf, __bf16* __restrict__ Cb,
          int M, int N, int K, int outBf16) {
  const int lane = threadIdx.x & 31;
  const int wid  = threadIdx.x >> 5;
  const int tilesN = N >> 6;
  const int wt = blockIdx.x * 8 + wid;
  const int tm = wt / tilesN;
  const int tn = (wt % tilesN) * 64;
  if (tm * 16 >= M) return;

  const __bf16* Arow = A + (size_t)(tm * 16 + (lane & 15)) * K + ((lane >> 4) << 3);
  f32x8 acc0 = {}, acc1 = {}, acc2 = {}, acc3 = {};
  for (int k = 0; k < K; k += 32) {
    bf16x16 a = lda_bf(Arow + k);
    acc0 = WMMA_BF16(a, ldb_w(W, K, tn +  0, k, lane), acc0);
    acc1 = WMMA_BF16(a, ldb_w(W, K, tn + 16, k, lane), acc1);
    acc2 = WMMA_BF16(a, ldb_w(W, K, tn + 32, k, lane), acc2);
    acc3 = WMMA_BF16(a, ldb_w(W, K, tn + 48, k, lane), acc3);
  }
  const int rowAdd = (lane >> 4) << 3;
  const int colB   = lane & 15;
  f32x8 accs[4] = {acc0, acc1, acc2, acc3};
#pragma unroll
  for (int t = 0; t < 4; ++t) {
    int col = tn + t * 16 + colB;
#pragma unroll
    for (int r = 0; r < 8; ++r) {
      size_t off = (size_t)(tm * 16 + rowAdd + r) * N + col;
      if (outBf16) Cb[off] = (__bf16)accs[t][r];
      else         Cf[off] = accs[t][r];
    }
  }
}

// Chunked causal taylor linear attention, one block per (b,h). 4 waves.
__global__ void __launch_bounds__(128)
attn_chunked(const float* __restrict__ qg, const float* __restrict__ kg,
             const __bf16* __restrict__ vg, const float* __restrict__ gw,
             __bf16* __restrict__ yn, float* __restrict__ kv_out) {
  __shared__ __attribute__((aligned(32))) __bf16 qf[CHUNK][FP_];     // 36 KB  features(q)
  __shared__ __attribute__((aligned(32))) __bf16 kf[CHUNK][FP_];     // 36 KB  features(k) [m][d]
  __shared__ __attribute__((aligned(32))) __bf16 kfT[FP_][CHUNK];    // 36 KB  features(k) [d][m]
  __shared__ __attribute__((aligned(32))) __bf16 vT[HD][CHUNK];      // 8 KB   v transposed [f][m]
  __shared__ __attribute__((aligned(32))) __bf16 Abuf[CHUNK][CHUNK]; // 8 KB   masked scores
  __shared__ __attribute__((aligned(32))) __bf16 Sbf[HD][FP_];       // 36 KB  bf16 mirror of S
  __shared__ float S[HD][FP_];                                       // 72 KB  running kv state
  __shared__ float ybuf[CHUNK][HD + 1];                              // 16.6 KB
  __shared__ float qld[2][CHUNK][FD];                                // 8 KB   TDM landing (q)
  __shared__ float kld[2][CHUNK][FD];                                // 8 KB   TDM landing (k)

  const int tid  = threadIdx.x;
  const int lane = tid & 31;
  const int wid  = tid >> 5;
  const int bi   = blockIdx.x >> 4;
  const int h    = blockIdx.x & 15;
  const int m0   = wid * 16;

  for (int i = tid; i < HD * FP_; i += 128) {
    (&S[0][0])[i] = 0.0f;
    (&Sbf[0][0])[i] = (__bf16)0.0f;
  }
  __syncthreads();

  const float c1 = 0.5f;                  // 1/d^(1/4), d=16
  const float c2 = 0.17677669529663687f;  // 1/(sqrt(d)*sqrt(2))

#if HAVE_TDM
  // prefetch chunk 0 via Tensor Data Mover (one issue per instruction; wave 0 only)
  if (wid == 0) {
    uint64_t qa = (uint64_t)(uintptr_t)qg + ((size_t)(bi * L_) * (NH * FD) + h * FD) * 4;
    uint64_t ka = (uint64_t)(uintptr_t)kg + ((size_t)(bi * L_) * (NH * FD) + h * FD) * 4;
    tdm_load_tile2d_f32((uint32_t)(uintptr_t)&qld[0][0][0], qa, FD, CHUNK, NH * FD);
    tdm_load_tile2d_f32((uint32_t)(uintptr_t)&kld[0][0][0], ka, FD, CHUNK, NH * FD);
  }
#endif

  for (int ci = 0; ci < NCH; ++ci) {
    const int r0  = ci * CHUNK;
    const int buf = ci & 1;

#if HAVE_TDM
    if (wid == 0) __builtin_amdgcn_s_wait_tensorcnt(0);
    __syncthreads();
    if (wid == 0 && (ci + 1) < NCH) {
      uint64_t qa = (uint64_t)(uintptr_t)qg +
                    ((size_t)(bi * L_ + r0 + CHUNK) * (NH * FD) + h * FD) * 4;
      uint64_t ka = (uint64_t)(uintptr_t)kg +
                    ((size_t)(bi * L_ + r0 + CHUNK) * (NH * FD) + h * FD) * 4;
      tdm_load_tile2d_f32((uint32_t)(uintptr_t)&qld[buf ^ 1][0][0], qa, FD, CHUNK, NH * FD);
      tdm_load_tile2d_f32((uint32_t)(uintptr_t)&kld[buf ^ 1][0][0], ka, FD, CHUNK, NH * FD);
    }
#else
    for (int idx = tid; idx < CHUNK * FD; idx += 128) {
      int n = idx >> 4, i = idx & 15;
      qld[buf][n][i] = qg[(size_t)(bi * L_ + r0 + n) * (NH * FD) + h * FD + i];
      kld[buf][n][i] = kg[(size_t)(bi * L_ + r0 + n) * (NH * FD) + h * FD + i];
    }
    __syncthreads();
#endif

    // ---- Phase A: taylor features (qf, kf, kfT) + v transpose into LDS ----
    {
      const int n = tid >> 1, sel = tid & 1;
      float qv[FD], kvv[FD];
#pragma unroll
      for (int i = 0; i < FD; ++i) { qv[i] = qld[buf][n][i]; kvv[i] = kld[buf][n][i]; }
      if (sel == 0) {
        qf[n][0] = (__bf16)1.0f;
        kf[n][0] = (__bf16)1.0f;  kfT[0][n] = (__bf16)1.0f;
#pragma unroll
        for (int i = 0; i < FD; ++i) {
          qf[n][1 + i] = (__bf16)(qv[i] * c1);
          __bf16 t = (__bf16)(kvv[i] * c1);
          kf[n][1 + i] = t;  kfT[1 + i][n] = t;
        }
        for (int d = FVAL; d < FP_; ++d) {
          qf[n][d] = (__bf16)0.0f;
          kf[n][d] = (__bf16)0.0f;  kfT[d][n] = (__bf16)0.0f;
        }
      }
#pragma unroll
      for (int i = sel * 8; i < sel * 8 + 8; ++i)
#pragma unroll
        for (int j = 0; j < FD; ++j) {
          qf[n][17 + i * FD + j] = (__bf16)(qv[i] * qv[j] * c2);
          __bf16 t = (__bf16)(kvv[i] * kvv[j] * c2);
          kf[n][17 + i * FD + j] = t;  kfT[17 + i * FD + j][n] = t;
        }
      for (int idx = tid; idx < CHUNK * HD; idx += 128) {
        int m = idx >> 6, f = idx & 63;
        vT[f][m] = vg[(size_t)(bi * L_ + r0 + m) * (NH * HD) + h * HD + f];
      }
    }
    __syncthreads();

    // ---- Phase B: inter-chunk  y = qf @ S^T  (B from bf16 mirror Sbf, contiguous) ----
    f32x8 y0 = {}, y1 = {}, y2 = {}, y3 = {};
    {
      const __bf16* arow = &qf[0][0] + (size_t)(m0 + (lane & 15)) * FP_ + ((lane >> 4) << 3);
#pragma unroll
      for (int k = 0; k < FP_; k += 32) {
        bf16x16 a = lda_bf(arow + k);
        y0 = WMMA_BF16(a, ldb_w(&Sbf[0][0], FP_,  0, k, lane), y0);
        y1 = WMMA_BF16(a, ldb_w(&Sbf[0][0], FP_, 16, k, lane), y1);
        y2 = WMMA_BF16(a, ldb_w(&Sbf[0][0], FP_, 32, k, lane), y2);
        y3 = WMMA_BF16(a, ldb_w(&Sbf[0][0], FP_, 48, k, lane), y3);
      }
    }

    // ---- Phase C: intra-chunk scores  A = tril(qf @ kf^T) ----
    {
      f32x8 a0 = {}, a1 = {}, a2 = {}, a3 = {};
      const __bf16* arow = &qf[0][0] + (size_t)(m0 + (lane & 15)) * FP_ + ((lane >> 4) << 3);
#pragma unroll
      for (int k = 0; k < FP_; k += 32) {
        bf16x16 a = lda_bf(arow + k);
        a0 = WMMA_BF16(a, ldb_w(&kf[0][0], FP_,  0, k, lane), a0);
        a1 = WMMA_BF16(a, ldb_w(&kf[0][0], FP_, 16, k, lane), a1);
        a2 = WMMA_BF16(a, ldb_w(&kf[0][0], FP_, 32, k, lane), a2);
        a3 = WMMA_BF16(a, ldb_w(&kf[0][0], FP_, 48, k, lane), a3);
      }
      const int colB = lane & 15, rowAdd = (lane >> 4) << 3;
      f32x8 accs[4] = {a0, a1, a2, a3};
#pragma unroll
      for (int t = 0; t < 4; ++t) {
        int col = t * 16 + colB;
#pragma unroll
        for (int r = 0; r < 8; ++r) {
          int rw = m0 + rowAdd + r;
          Abuf[rw][col] = (__bf16)((col <= rw) ? accs[t][r] : 0.0f);
        }
      }
    }

    // ---- Phase D: y += A @ v  (B from vT, weight-style [f][m]) ----
    {
      const __bf16* arow = &Abuf[0][0] + (size_t)(m0 + (lane & 15)) * CHUNK + ((lane >> 4) << 3);
#pragma unroll
      for (int k = 0; k < CHUNK; k += 32) {
        bf16x16 a = lda_bf(arow + k);
        y0 = WMMA_BF16(a, ldb_w(&vT[0][0], CHUNK,  0, k, lane), y0);
        y1 = WMMA_BF16(a, ldb_w(&vT[0][0], CHUNK, 16, k, lane), y1);
        y2 = WMMA_BF16(a, ldb_w(&vT[0][0], CHUNK, 32, k, lane), y2);
        y3 = WMMA_BF16(a, ldb_w(&vT[0][0], CHUNK, 48, k, lane), y3);
      }
    }
    __syncthreads();  // all waves done reading Sbf (phase B) before update

    // ---- Phase E: S[f,d] += (vT @ kf)[f,d], per-wave f-strip; refresh bf16 mirror ----
    {
      const __bf16* arow = &vT[0][0] + (size_t)(m0 + (lane & 15)) * CHUNK + ((lane >> 4) << 3);
      const int colB = lane & 15, rowAdd = (lane >> 4) << 3;
#pragma unroll 2
      for (int dt = 0; dt < FP_ / 16; ++dt) {
        f32x8 s = {};
#pragma unroll
        for (int k = 0; k < CHUNK; k += 32) {
          bf16x16 a = lda_bf(arow + k);
          s = WMMA_BF16(a, ldb_w(&kfT[0][0], CHUNK, dt * 16, k, lane), s);
        }
        int col = dt * 16 + colB;
#pragma unroll
        for (int r = 0; r < 8; ++r) {
          int row = m0 + rowAdd + r;
          float nv = S[row][col] + s[r];
          S[row][col] = nv;
          Sbf[row][col] = (__bf16)nv;
        }
      }
    }

    // ---- Phase F: RMSNorm + store bf16 y ----
    {
      const int colB = lane & 15, rowAdd = (lane >> 4) << 3;
      f32x8 ys[4] = {y0, y1, y2, y3};
#pragma unroll
      for (int t = 0; t < 4; ++t)
#pragma unroll
        for (int r = 0; r < 8; ++r)
          ybuf[m0 + rowAdd + r][t * 16 + colB] = ys[t][r];
    }
    __syncthreads();
    if (tid < CHUNK) {
      float ss = 0.0f;
#pragma unroll
      for (int f = 0; f < HD; ++f) { float v = ybuf[tid][f]; ss += v * v; }
      float rms = rsqrtf(ss * (1.0f / HD) + 1e-5f);
      __bf16* outp = yn + (size_t)(bi * L_ + r0 + tid) * (NH * HD) + h * HD;
#pragma unroll
      for (int f = 0; f < HD; ++f) outp[f] = (__bf16)(ybuf[tid][f] * rms * gw[f]);
    }
    __syncthreads();
  }

  // ---- write kv_state [b,h,1,64,273] ----
  float* kvp = kv_out + (size_t)(bi * NH + h) * HD * FVAL;
  for (int idx = tid; idx < HD * FVAL; idx += 128) {
    int f = idx / FVAL, d = idx - f * FVAL;
    kvp[idx] = S[f][d];
  }
}

// ---------------- host launch ----------------
extern "C" void kernel_launch(void* const* d_in, const int* in_sizes, int n_in,
                              void* d_out, int out_size, void* d_ws, size_t ws_size,
                              hipStream_t stream) {
  (void)in_sizes; (void)n_in; (void)out_size; (void)ws_size;
  const float* hs = (const float*)d_in[0];
  const float* Wq = (const float*)d_in[1];
  const float* Wk = (const float*)d_in[2];
  const float* Wv = (const float*)d_in[3];
  const float* Wo = (const float*)d_in[4];
  const float* gw = (const float*)d_in[5];

  float* out = (float*)d_out;
  float* kv  = out + (size_t)B_ * L_ * DM;

  char* ws = (char*)d_ws;
  size_t off = 0;
  auto alloc = [&](size_t bytes) -> void* {
    void* p = ws + off;
    off += (bytes + 255) & ~(size_t)255;
    return p;
  };
  const int M = B_ * L_;
  __bf16* xbf  = (__bf16*)alloc((size_t)M * DM * 2);
  __bf16* Wq_b = (__bf16*)alloc((size_t)NH * FD * DM * 2);
  __bf16* Wk_b = (__bf16*)alloc((size_t)NH * FD * DM * 2);
  __bf16* Wv_b = (__bf16*)alloc((size_t)DM * DM * 2);
  __bf16* Wo_b = (__bf16*)alloc((size_t)DM * DM * 2);
  float*  q32  = (float*)alloc((size_t)M * NH * FD * 4);
  float*  k32  = (float*)alloc((size_t)M * NH * FD * 4);
  __bf16* vbf  = (__bf16*)alloc((size_t)M * DM * 2);
  __bf16* ynb  = (__bf16*)alloc((size_t)M * DM * 2);

  auto cvt = [&](const float* s, __bf16* d, int n) {
    cvt_f32_bf16<<<(n + 255) / 256, 256, 0, stream>>>(s, d, n);
  };
  cvt(hs, xbf, M * DM);
  cvt(Wq, Wq_b, NH * FD * DM);
  cvt(Wk, Wk_b, NH * FD * DM);
  cvt(Wv, Wv_b, DM * DM);
  cvt(Wo, Wo_b, DM * DM);

  // q,k projections: [M,1024] x [256,1024]^T -> fp32 [M,256]
  {
    int blocks = (M / 16) * ((NH * FD) / 64) / 8;
    gemm_wmma<<<blocks, 256, 0, stream>>>(xbf, Wq_b, q32, nullptr, M, NH * FD, DM, 0);
    gemm_wmma<<<blocks, 256, 0, stream>>>(xbf, Wk_b, k32, nullptr, M, NH * FD, DM, 0);
  }
  // v projection: bf16 [M,1024]
  {
    int blocks = (M / 16) * (DM / 64) / 8;
    gemm_wmma<<<blocks, 256, 0, stream>>>(xbf, Wv_b, nullptr, vbf, M, DM, DM, 1);
  }
  // chunked attention, one block per (b,h)
  attn_chunked<<<B_ * NH, 128, 0, stream>>>(q32, k32, vbf, gw, ynb, kv);
  // output projection -> fp32 d_out
  {
    int blocks = (M / 16) * (DM / 64) / 8;
    gemm_wmma<<<blocks, 256, 0, stream>>>(ynb, Wo_b, out, nullptr, M, DM, DM, 0);
  }
}
